// BuildingBlockEmbedder_55387898249705
// MI455X (gfx1250) — matrix operand order repaired and checked by reading, not compiled, not run on hardware
//
#include <hip/hip_runtime.h>
#include <hip/hip_bf16.h>

// EGNN BuildingBlockEmbedder for gfx1250 (MI455X), wave32 + WMMA f16.
//
// Pipeline (all on `stream`):
//   k_embed  : h0 = emb[atom_types-1]                       (f32, ws)
//   k_geo    : per-edge [radial, 50 gaussians, pad] -> f16 K=64 rows
//   k_w_*    : transpose weights to N-major f16 for 32B B-fragment loads
//   k_layer  : (x4) per-BB fused layer:
//                PQ = h_bb @ [W1a|W1b]            (WMMA)
//                msg1 = relu(geo@W1c + P[r]+Q[c]+b1)   (WMMA, W1c staged in LDS)
//                msg2 = relu(msg1@W2 + b2)        (WMMA, W2 staged in LDS),
//                row-grouped ds_add_f32 scatter into agg
//                t = relu([h|agg]@NW1+nb1); upd=t@NW2+nb2; h = 2h+upd (WMMA)
//   k_pool   : mean over 32 atoms per BB -> d_out

typedef __attribute__((ext_vector_type(16))) _Float16 v16h;
typedef __attribute__((ext_vector_type(8)))  _Float16 v8h;
typedef __attribute__((ext_vector_type(8)))  float    v8f;

#define BN     4096      // B*N nodes
#define NBB    128       // building blocks (B*M_BB)
#define NPB    32        // nodes per BB
#define EPB    992       // edges per BB (32*31)
#define ETOT   126976    // NBB*EPB
#define CN     128       // node channels
#define CH     128       // hidden channels
#define CE     50        // edge gaussians
#define KGEO   64        // padded geo K
#define NL     4         // layers

// ---------------- WMMA fragment helpers (CDNA5 wave32 layouts) --------------
// A: 16x32 f16, row-major source with leading dim `ld`.
// lane L: M = L&15, half = L>>4; a[0..7] = K(8*half .. +7), a[8..15] = K(16+8*half .. +7)
__device__ __forceinline__ v16h load_a_frag(const _Float16* A, int ld, int k0, int lane) {
  int m = lane & 15, hh = lane >> 4;
  const _Float16* p = A + m * ld + k0 + 8 * hh;
  v8h lo = *(const v8h*)p;
  v8h hi = *(const v8h*)(p + 16);
  v16h r;
#pragma unroll
  for (int i = 0; i < 8; ++i) { r[i] = lo[i]; r[i + 8] = hi[i]; }
  return r;
}

// B: 32x16 f16 from N-major (transposed) weights Wt[n][k], leading dim ldk.
// lane L: N = L&15, half = L>>4; b[0..7] = K(k0 + 8*half .. +7), b[8..15] = K(k0 + 16 + 8*half .. +7)
__device__ __forceinline__ v16h load_b_frag(const _Float16* Wt, int ldk, int n0, int k0, int lane) {
  int n = lane & 15, hh = lane >> 4;
  const _Float16* p = Wt + (size_t)(n0 + n) * ldk + k0 + 8 * hh;
  v8h lo = *(const v8h*)p;
  v8h hi = *(const v8h*)(p + 16);
  v16h r;
#pragma unroll
  for (int i = 0; i < 8; ++i) { r[i] = lo[i]; r[i + 8] = hi[i]; }
  return r;
}

// C/D: c[r] <-> (M = r + 8*(lane>=16), N = lane&15)
__device__ __forceinline__ v8f wmma16(v16h a, v16h b, v8f c) {
  return __builtin_amdgcn_wmma_f32_16x16x32_f16(false, a, false, b, (short)0, c, false, false);
}

// ------------------------------- prep kernels -------------------------------
__global__ void k_embed(const int* __restrict__ at, const float* __restrict__ emb,
                        float* __restrict__ h) {
  int idx = blockIdx.x * blockDim.x + threadIdx.x;
  if (idx >= BN * CN) return;
  int v = idx / CN, c = idx - v * CN;
  h[idx] = emb[(at[v] - 1) * CN + c];
}

__global__ void k_geo(const float* __restrict__ x, const int* __restrict__ row,
                      const int* __restrict__ col, _Float16* __restrict__ geo) {
  int e = blockIdx.x * blockDim.x + threadIdx.x;
  if (e >= ETOT) return;
  int i = row[e], j = col[e];
  float dx = x[3 * i]     - x[3 * j];
  float dy = x[3 * i + 1] - x[3 * j + 1];
  float dz = x[3 * i + 2] - x[3 * j + 2];
  float d2 = dx * dx + dy * dy + dz * dz;   // Angstrom^2
  float d  = sqrtf(d2);
  _Float16* g = geo + (size_t)e * KGEO;
  g[0] = (_Float16)(d2 * 0.01f);            // radial = (0.1*d)^2 in nm^2
  const float step  = 5.0f / 49.0f;
  const float coeff = -0.5f / (step * step);
#pragma unroll 1
  for (int k = 0; k < CE; ++k) {
    float t = d - step * (float)k;
    g[1 + k] = (_Float16)__expf(coeff * t * t);
  }
#pragma unroll
  for (int k = CE + 1; k < KGEO; ++k) g[k] = (_Float16)0.f;
}

// edge_w1 rows 0..255 -> [NL][256 n][128 k] f16 (n<128 uses W1a rows, else W1b)
__global__ void k_w_ab(const float* __restrict__ w1, _Float16* __restrict__ o) {
  int idx = blockIdx.x * blockDim.x + threadIdx.x;
  if (idx >= NL * 256 * 128) return;
  int l = idx / (256 * 128), r = idx - l * 256 * 128;
  int n = r / 128, k = r - n * 128;
  int srcRow = (n < 128) ? k : (128 + k);
  o[idx] = (_Float16)w1[((size_t)l * 307 + srcRow) * 128 + (n & 127)];
}

// edge_w1 rows 256..306 (radial + edge_attr) -> [NL][128 n][64 k] f16, zero-padded
__global__ void k_w_c(const float* __restrict__ w1, _Float16* __restrict__ o) {
  int idx = blockIdx.x * blockDim.x + threadIdx.x;
  if (idx >= NL * 128 * 64) return;
  int l = idx / (128 * 64), r = idx - l * 128 * 64;
  int n = r / 64, k = r - n * 64;
  o[idx] = (k < 51) ? (_Float16)w1[((size_t)l * 307 + 256 + k) * 128 + n] : (_Float16)0.f;
}

// generic [NL][K][Nc] -> [NL][Nc][K] f16
__global__ void k_w_t(const float* __restrict__ w, _Float16* __restrict__ o, int K, int Nc) {
  int idx = blockIdx.x * blockDim.x + threadIdx.x;
  if (idx >= NL * K * Nc) return;
  int l = idx / (K * Nc), r = idx - l * K * Nc;
  int n = r / K, k = r - n * K;
  o[idx] = (_Float16)w[((size_t)l * K + k) * Nc + n];
}

// ------------------------------ fused layer ---------------------------------
// Dynamic LDS (136 KB):
//   h16  [32][128] f16   8 KB
//   PQ   [32][256] f32  32 KB
//   agg  [32][128] f32  16 KB
//   w1cl [128][64] f16  16 KB   (N-major, staged once per block)
//   w2l  [128][128] f16 32 KB   (N-major, staged once per block)
//   dyn  32 KB: edge phase = per-wave msg1 [16][128] f16; node phase = ni16+t16
__global__ __launch_bounds__(256) void k_layer(
    float* __restrict__ h, const _Float16* __restrict__ geo,
    const _Float16* __restrict__ w1ab, const _Float16* __restrict__ w1c,
    const _Float16* __restrict__ w2,   const _Float16* __restrict__ nw1,
    const _Float16* __restrict__ nw2,
    const float* __restrict__ eb1, const float* __restrict__ eb2,
    const float* __restrict__ nb1, const float* __restrict__ nb2)
{
  extern __shared__ __align__(32) char smem[];
  _Float16* h16  = (_Float16*)smem;                      //  8192 B
  float*    PQ   = (float*)(smem + 8192);                // 32768 B
  float*    agg  = (float*)(smem + 40960);               // 16384 B
  _Float16* w1cl = (_Float16*)(smem + 57344);            // 16384 B
  _Float16* w2l  = (_Float16*)(smem + 73728);            // 32768 B
  char*     dyn  = smem + 106496;                        // 32768 B (phase-shared)
  _Float16* msg1 = (_Float16*)dyn;                       // edge phase
  _Float16* ni16 = (_Float16*)dyn;                       // node phase: [32][256]
  _Float16* t16  = (_Float16*)(dyn + 16384);             // node phase: [32][128]

  const int tid = threadIdx.x, lane = tid & 31, wave = tid >> 5;
  const int nloc = lane & 15, halfl = lane >> 4;
  const int bb = blockIdx.x, vb = bb * NPB;

  for (int idx = tid; idx < NPB * CN; idx += 256) {
    h16[idx] = (_Float16)h[(size_t)vb * CN + idx];
    agg[idx] = 0.f;
  }
  // stage W1c^T (128x64) and W2^T (128x128) into LDS, 16B vectors
  for (int idx = tid * 8; idx < 128 * 64; idx += 256 * 8)
    *(v8h*)(w1cl + idx) = *(const v8h*)(w1c + idx);
  for (int idx = tid * 8; idx < 128 * 128; idx += 256 * 8)
    *(v8h*)(w2l + idx) = *(const v8h*)(w2 + idx);
  __syncthreads();

  // -------- PQ = h_bb @ [W1a | W1b]  (32 x 256, K=128) --------
  for (int tt = wave; tt < 32; tt += 8) {
    int mt = tt >> 4, nt = tt & 15;
    v8f c = {};
#pragma unroll
    for (int kk = 0; kk < 4; ++kk) {
      v16h a = load_a_frag(h16 + mt * 16 * CN, CN, kk * 32, lane);
      v16h b = load_b_frag(w1ab, CN, nt * 16, kk * 32, lane);
      c = wmma16(a, b, c);
    }
    int n = nt * 16 + nloc, mrow = mt * 16 + 8 * halfl;
#pragma unroll
    for (int r = 0; r < 8; ++r) PQ[(mrow + r) * 256 + n] = c[r];
  }
  __syncthreads();

  // -------- edge phase: 62 tiles of 16 edges --------
  float bias1[8], bias2[8];
#pragma unroll
  for (int nt = 0; nt < 8; ++nt) {
    bias1[nt] = eb1[nt * 16 + nloc];
    bias2[nt] = eb2[nt * 16 + nloc];
  }
  _Float16* mymsg = msg1 + wave * 16 * CH;
  for (int t = wave; t < 62; t += 8) {
    int e0 = t * 16;
    const _Float16* ga = geo + ((size_t)bb * EPB + e0) * KGEO;
    v16h ga0 = load_a_frag(ga, KGEO, 0, lane);    // hoisted: reused by all 8 nt
    v16h ga1 = load_a_frag(ga, KGEO, 32, lane);
    int irow[8], jcol[8];
#pragma unroll
    for (int r = 0; r < 8; ++r) {
      int le = e0 + r + 8 * halfl;            // BB-local edge id, row-sorted
      int i = le / 31, jj = le - i * 31;
      irow[r] = i;
      jcol[r] = jj + (jj >= i ? 1 : 0);
    }
    // msg1 = relu(geo @ W1c + P[row] + Q[col] + b1)
#pragma unroll
    for (int nt = 0; nt < 8; ++nt) {
      v8f c = {};
      c = wmma16(ga0, load_b_frag(w1cl, KGEO, nt * 16, 0, lane), c);
      c = wmma16(ga1, load_b_frag(w1cl, KGEO, nt * 16, 32, lane), c);
      int n = nt * 16 + nloc;
#pragma unroll
      for (int r = 0; r < 8; ++r) {
        float v = c[r] + PQ[irow[r] * 256 + n] + PQ[jcol[r] * 256 + 128 + n] + bias1[nt];
        mymsg[(r + 8 * halfl) * CH + n] = (_Float16)(v > 0.f ? v : 0.f);
      }
    }
    // msg2 = relu(msg1 @ W2 + b2); agg[row] += msg2 (row-grouped scatter)
    v16h am[4];
#pragma unroll
    for (int kk = 0; kk < 4; ++kk) am[kk] = load_a_frag(mymsg, CH, kk * 32, lane);
#pragma unroll
    for (int nt = 0; nt < 8; ++nt) {
      v8f c = {};
#pragma unroll
      for (int kk = 0; kk < 4; ++kk)
        c = wmma16(am[kk], load_b_frag(w2l, CH, nt * 16, kk * 32, lane), c);
      int n = nt * 16 + nloc;
      float acc = 0.f;
      int cur = irow[0];
#pragma unroll
      for (int r = 0; r < 8; ++r) {
        float v = c[r] + bias2[nt];
        v = v > 0.f ? v : 0.f;
        if (irow[r] != cur) {                  // <=2 distinct rows per 8 edges
          atomicAdd(&agg[cur * CH + n], acc);  // ds_add_f32
          cur = irow[r];
          acc = 0.f;
        }
        acc += v;
      }
      atomicAdd(&agg[cur * CH + n], acc);
    }
  }
  __syncthreads();

  // -------- node phase: ni = [h | agg] --------
  for (int idx = tid; idx < NPB * CN; idx += 256) {
    int v = idx / CN, c = idx - v * CN;
    ni16[v * 256 + c]       = h16[idx];
    ni16[v * 256 + 128 + c] = (_Float16)agg[idx];
  }
  __syncthreads();

  // t = relu(ni @ NW1 + nb1)   (32 x 128, K=256)
  for (int tt = wave; tt < 16; tt += 8) {
    int mt = tt >> 3, nt = tt & 7;
    v8f c = {};
#pragma unroll
    for (int kk = 0; kk < 8; ++kk) {
      v16h a = load_a_frag(ni16 + mt * 16 * 256, 256, kk * 32, lane);
      v16h b = load_b_frag(nw1, 256, nt * 16, kk * 32, lane);
      c = wmma16(a, b, c);
    }
    int n = nt * 16 + nloc, mrow = mt * 16 + 8 * halfl;
    float bias = nb1[n];
#pragma unroll
    for (int r = 0; r < 8; ++r) {
      float v = c[r] + bias;
      t16[(mrow + r) * CH + n] = (_Float16)(v > 0.f ? v : 0.f);
    }
  }
  __syncthreads();

  // upd = t @ NW2 + nb2 ; h = 2h + upd
  for (int tt = wave; tt < 16; tt += 8) {
    int mt = tt >> 3, nt = tt & 7;
    v8f c = {};
#pragma unroll
    for (int kk = 0; kk < 4; ++kk) {
      v16h a = load_a_frag(t16 + mt * 16 * CH, CH, kk * 32, lane);
      v16h b = load_b_frag(nw2, CH, nt * 16, kk * 32, lane);
      c = wmma16(a, b, c);
    }
    int n = nt * 16 + nloc, mrow = mt * 16 + 8 * halfl;
    float bias = nb2[n];
#pragma unroll
    for (int r = 0; r < 8; ++r) {
      size_t gi = (size_t)(vb + mrow + r) * CN + n;
      h[gi] = 2.f * h[gi] + c[r] + bias;
    }
  }
}

__global__ void k_pool(const float* __restrict__ h, float* __restrict__ out) {
  int bb = blockIdx.x, c = threadIdx.x;
  float s = 0.f;
#pragma unroll
  for (int v = 0; v < NPB; ++v) s += h[((size_t)bb * NPB + v) * CN + c];
  out[bb * CN + c] = s * (1.0f / NPB);
}

// ------------------------------- launcher -----------------------------------
extern "C" void kernel_launch(void* const* d_in, const int* in_sizes, int n_in,
                              void* d_out, int out_size, void* d_ws, size_t ws_size,
                              hipStream_t stream) {
  const int*   atom_types = (const int*)  d_in[0];
  const float* coords     = (const float*)d_in[1];
  const int*   row        = (const int*)  d_in[2];
  const int*   col        = (const int*)  d_in[3];
  const float* emb        = (const float*)d_in[4];
  const float* edge_w1    = (const float*)d_in[5];
  const float* edge_b1    = (const float*)d_in[6];
  const float* edge_w2    = (const float*)d_in[7];
  const float* edge_b2    = (const float*)d_in[8];
  const float* node_w1    = (const float*)d_in[9];
  const float* node_b1    = (const float*)d_in[10];
  const float* node_w2    = (const float*)d_in[11];
  const float* node_b2    = (const float*)d_in[12];
  (void)in_sizes; (void)n_in; (void)out_size; (void)ws_size;

  char* ws = (char*)d_ws;
  float*    h   = (float*)ws;                                   // 4096*128*4 = 2 MB
  size_t    off = (size_t)BN * CN * 4;
  _Float16* geo = (_Float16*)(ws + off);  off += (size_t)ETOT * KGEO * 2;   // 16.25 MB
  _Float16* w1ab = (_Float16*)(ws + off); off += (size_t)NL * 256 * 128 * 2;
  _Float16* w1c  = (_Float16*)(ws + off); off += (size_t)NL * 128 * 64 * 2;
  _Float16* w2   = (_Float16*)(ws + off); off += (size_t)NL * 128 * 128 * 2;
  _Float16* nw1  = (_Float16*)(ws + off); off += (size_t)NL * 128 * 256 * 2;
  _Float16* nw2  = (_Float16*)(ws + off); off += (size_t)NL * 128 * 128 * 2;

  const size_t SMEM = 8192 + 32768 + 16384 + 16384 + 32768 + 32768;  // 136 KB
  (void)hipFuncSetAttribute((const void*)k_layer,
                            hipFuncAttributeMaxDynamicSharedMemorySize, (int)SMEM);

  k_embed<<<(BN * CN + 255) / 256, 256, 0, stream>>>(atom_types, emb, h);
  k_geo  <<<(ETOT + 127) / 128, 128, 0, stream>>>(coords, row, col, geo);
  k_w_ab <<<(NL * 256 * 128 + 255) / 256, 256, 0, stream>>>(edge_w1, w1ab);
  k_w_c  <<<(NL * 128 * 64  + 255) / 256, 256, 0, stream>>>(edge_w1, w1c);
  k_w_t  <<<(NL * 128 * 128 + 255) / 256, 256, 0, stream>>>(edge_w2, w2, 128, 128);
  k_w_t  <<<(NL * 256 * 128 + 255) / 256, 256, 0, stream>>>(node_w1, nw1, 256, 128);
  k_w_t  <<<(NL * 128 * 128 + 255) / 256, 256, 0, stream>>>(node_w2, nw2, 128, 128);

  for (int l = 0; l < NL; ++l) {
    k_layer<<<NBB, 256, SMEM, stream>>>(
        h, geo,
        w1ab + (size_t)l * 256 * 128, w1c + (size_t)l * 128 * 64,
        w2   + (size_t)l * 128 * 128, nw1 + (size_t)l * 128 * 256,
        nw2  + (size_t)l * 128 * 128,
        edge_b1 + l * 128, edge_b2 + l * 128,
        node_b1 + l * 128, node_b2 + l * 128);
  }
  k_pool<<<NBB, CN, 0, stream>>>(h, (float*)d_out);
}
